// Shallow_GMUCNN_57655640982276
// MI455X (gfx1250) — matrix operand order repaired and checked by reading, not compiled, not run on hardware
//
#include <hip/hip_runtime.h>
#include <hip/hip_bf16.h>
#include <stdint.h>

// ---------------------------------------------------------------------------
// Shallow_GMUCNN for MI455X (gfx1250), wave32 + bf16 WMMA (16x16x32, f32 acc).
// SRN layer reduced algebraically: err = (||y||^2 - v^T (X^T X)^-1 v)/(n*var),
// v = X^T y computed as a 256 x 96 x 3600 GEMM per batch image.
// Convs are im2col GEMMs. Conv biases cancel under training-mode BN.
// GEMM operands are fragment-swizzled (contiguous 32B per lane-fragment) and
// each wave computes a 64x16 C column: per K-step 1 B + 4 A vector loads feed
// 4 v_wmma ops (B-fragment reuse, ~2x arithmetic intensity vs 16x16 tiles).
// ---------------------------------------------------------------------------

typedef __bf16 bf16_t;
typedef __bf16 v16bf __attribute__((ext_vector_type(16)));
typedef float  v8f   __attribute__((ext_vector_type(8)));

#define BATCH   4
#define BN_EPS  1e-5f
#define EPS_NOISE 1e-3f

__device__ __forceinline__ bf16_t f2bf(float f) {
  uint32_t u = __builtin_bit_cast(uint32_t, f);
  uint32_t r = u + 0x7FFFu + ((u >> 16) & 1u);          // round-to-nearest-even
  uint16_t h = (uint16_t)(r >> 16);
  return __builtin_bit_cast(bf16_t, h);
}

// Fragment-swizzled element offsets (bf16 element index into packed buffer).
// A (M x K): GEMM reads v16bf at ((mT*(K/32)+kt)*32 + lane); lane&15 = row,
// lane>>4 = K half; element e = k mod 16 (k = 16*half + e).
__device__ __forceinline__ size_t aoff(int m, int k, int K) {
  int kk = k & 31;
  int half = kk >> 4;
  int e = kk & 15;
  return ((size_t)((m >> 4) * (K >> 5) + (k >> 5)) * 32 +
          (m & 15) + (half << 4)) * 16 + e;
}
// B (K x N): GEMM reads v16bf at ((kt*(N/16)+nT)*32 + lane); lane&15 = col,
// lane>>4 = K half; element e = k mod 16.
__device__ __forceinline__ size_t boff(int k, int n, int N) {
  return ((size_t)((k >> 5) * (N >> 4) + (n >> 4)) * 32 +
          (((k & 31) >> 4) << 4) + (n & 15)) * 16 + (k & 15);
}

__device__ __forceinline__ uint32_t hash_u32(uint32_t s) {
  s ^= s >> 17; s *= 0xed5ad4bbu;
  s ^= s >> 11; s *= 0xac4c1b51u;
  s ^= s >> 15; s *= 0x31848babu;
  s ^= s >> 14;
  return s;
}

__device__ __forceinline__ float gaussian(uint32_t seed) {
  uint32_t a = hash_u32(seed * 2654435761u + 0x9e3779b9u);
  uint32_t b = hash_u32(seed ^ 0x85ebca6bu);
  float u1 = (float)(a >> 8) * (1.0f / 16777216.0f) + 1e-7f;
  float u2 = (float)(b >> 8) * (1.0f / 16777216.0f);
  return sqrtf(-2.0f * logf(u1)) * cosf(6.28318530718f * u2);
}

// ---------------------------------------------------------------------------
// bf16 GEMM on fragment-packed operands. C[M,N] f32 row-major.
// One wave32 per 64x16 C column (4 M-tiles); M must be a multiple of 64.
// Per K-step: 1 B-fragment + 4 A-fragment vector loads -> 4 v_wmma.
// blockIdx.z = batch (strides in v16bf units for B, floats for C).
// ---------------------------------------------------------------------------
__global__ void k_gemm_bf16(const v16bf* __restrict__ Ap,
                            const v16bf* __restrict__ Bp,
                            float* __restrict__ C,
                            int M, int N, int K,
                            long long strideBv, long long strideC) {
  const int nT = blockIdx.x, mT0 = blockIdx.y * 4, bz = blockIdx.z;
  const int lane = threadIdx.x & 31;
  const int kTiles = K >> 5;
  const size_t bstep = (size_t)(N >> 4) * 32;   // v16bf per K-tile row of B
  const size_t astep = (size_t)kTiles * 32;     // v16bf per M-tile of A

  const v16bf* A0    = Ap + (size_t)mT0 * astep + lane;
  const v16bf* Bbase = Bp + (size_t)bz * strideBv + (size_t)nT * 32 + lane;

  v8f acc0 = {0.f, 0.f, 0.f, 0.f, 0.f, 0.f, 0.f, 0.f};
  v8f acc1 = acc0, acc2 = acc0, acc3 = acc0;

#pragma unroll 2
  for (int kt = 0; kt < kTiles; ++kt) {
    __builtin_prefetch(Bbase + (size_t)(kt + 2) * bstep, 0, 0);
    __builtin_prefetch(A0 + (size_t)(kt + 2) * 32, 0, 0);
    v16bf b  = Bbase[(size_t)kt * bstep];
    v16bf a0 = A0[(size_t)kt * 32];
    v16bf a1 = A0[astep + (size_t)kt * 32];
    v16bf a2 = A0[2 * astep + (size_t)kt * 32];
    v16bf a3 = A0[3 * astep + (size_t)kt * 32];
    acc0 = __builtin_amdgcn_wmma_f32_16x16x32_bf16(
        false, a0, false, b, (short)0, acc0, false, false);
    acc1 = __builtin_amdgcn_wmma_f32_16x16x32_bf16(
        false, a1, false, b, (short)0, acc1, false, false);
    acc2 = __builtin_amdgcn_wmma_f32_16x16x32_bf16(
        false, a2, false, b, (short)0, acc2, false, false);
    acc3 = __builtin_amdgcn_wmma_f32_16x16x32_bf16(
        false, a3, false, b, (short)0, acc3, false, false);
  }

  const int half = lane >> 4, l15 = lane & 15;
  float* Cp = C + (long long)bz * strideC;
  const int n = nT * 16 + l15;
#pragma unroll
  for (int r = 0; r < 8; ++r)         // C/D: VGPR r -> M = r + 8*half
    Cp[(size_t)((mT0 + 0) * 16 + r + 8 * half) * N + n] = acc0[r];
#pragma unroll
  for (int r = 0; r < 8; ++r)
    Cp[(size_t)((mT0 + 1) * 16 + r + 8 * half) * N + n] = acc1[r];
#pragma unroll
  for (int r = 0; r < 8; ++r)
    Cp[(size_t)((mT0 + 2) * 16 + r + 8 * half) * N + n] = acc2[r];
#pragma unroll
  for (int r = 0; r < 8; ++r)
    Cp[(size_t)((mT0 + 3) * 16 + r + 8 * half) * N + n] = acc3[r];
}

// ---------------------------------------------------------------------------
// Build Xe (packed bf16 A, 256x96: rows o*4+f, ones column f=0, zero-padded)
// and Xinv = (X^T X)^-1 per output channel (64 x 4x4) via Gauss-Jordan.
// srn_w is (64, 3, 5, 5, 3) -> contiguous (o, n, s) with n = c*25+i*5+j.
// ---------------------------------------------------------------------------
__global__ void k_prep_x(const float* __restrict__ srn_w,
                         bf16_t* __restrict__ Xe, float* __restrict__ Xinv) {
  int t = threadIdx.x;                     // 256 threads, one Xe row each
  {
    int o = t >> 2, f = t & 3;
    for (int n = 0; n < 96; ++n) {
      float v = 0.f;
      if (n < 75) v = (f == 0) ? 1.0f : srn_w[(o * 75 + n) * 3 + (f - 1)];
      Xe[aoff(t, n, 96)] = f2bf(v);
    }
  }
  if (t < 64) {
    int o = t;
    float a[4][8];
    for (int f = 0; f < 4; ++f)
      for (int j = 0; j < 8; ++j) a[f][j] = (j == f + 4) ? 1.f : 0.f;
    for (int n = 0; n < 75; ++n) {
      float xv[4];
      xv[0] = 1.f;
      for (int s = 0; s < 3; ++s) xv[s + 1] = srn_w[(o * 75 + n) * 3 + s];
      for (int f = 0; f < 4; ++f)
        for (int g = 0; g < 4; ++g) a[f][g] += xv[f] * xv[g];
    }
    for (int it = 0; it < 4; ++it) {
      float inv = 1.0f / a[it][it];
      for (int j = 0; j < 8; ++j) a[it][j] *= inv;
      for (int r = 0; r < 4; ++r)
        if (r != it) {
          float fct = a[r][it];
          for (int j = 0; j < 8; ++j) a[r][j] -= fct * a[it][j];
        }
    }
    for (int f = 0; f < 4; ++f)
      for (int g = 0; g < 4; ++g) Xinv[o * 16 + f * 4 + g] = a[f][4 + g];
  }
}

// ---------------------------------------------------------------------------
// Unfold (k=5) + additive noise -> packed bf16 y[B][K=96][N=3600]
// (rows 75..95 zero) and per-column sum of squares ssq[B,3600] in f32.
// ---------------------------------------------------------------------------
__global__ void k_unfold(const float* __restrict__ x, bf16_t* __restrict__ ybf,
                         float* __restrict__ ssq) {
  int idx = blockIdx.x * blockDim.x + threadIdx.x;
  if (idx >= BATCH * 3600) return;
  int b = idx / 3600, l = idx % 3600;
  int ho = l / 60, wo = l % 60;
  bf16_t* yb = ybf + (size_t)b * 96 * 3600;
  float acc = 0.f;
  int n = 0;
  for (int c = 0; c < 3; ++c)
    for (int i = 0; i < 5; ++i)
      for (int j = 0; j < 5; ++j, ++n) {
        float v = x[((b * 3 + c) * 64 + ho + i) * 64 + wo + j];
        v += EPS_NOISE * gaussian((uint32_t)((b * 96 + n) * 3600 + l));
        acc += v * v;
        yb[boff(n, l, 3600)] = f2bf(v);
      }
  for (; n < 96; ++n) yb[boff(n, l, 3600)] = f2bf(0.f);
  ssq[idx] = acc;
}

// ---------------------------------------------------------------------------
// SRN postprocess: err -> normalized activation, fused with 2x2 avg-pool.
// V is [B,256,3600] (rows o*4+f). Output h1p (B,64,30,30).
// ---------------------------------------------------------------------------
__global__ void k_srn_post(const float* __restrict__ V,
                           const float* __restrict__ ssq,
                           const float* __restrict__ Xinv,
                           float* __restrict__ h1p) {
  int idx = blockIdx.x * blockDim.x + threadIdx.x;
  if (idx >= BATCH * 64 * 900) return;
  int ow = idx % 30; int t2 = idx / 30;
  int oh = t2 % 30;  t2 /= 30;
  int o  = t2 % 64;  int b = t2 / 64;
  float xi[16];
#pragma unroll
  for (int q = 0; q < 16; ++q) xi[q] = Xinv[o * 16 + q];
  const float c1 = 0.3678794411714423f;  // e^-1
  float accp = 0.f;
  for (int dy = 0; dy < 2; ++dy)
    for (int dx = 0; dx < 2; ++dx) {
      int l = (2 * oh + dy) * 60 + (2 * ow + dx);
      size_t base = ((size_t)b * 256 + o * 4) * 3600 + l;
      float v0 = V[base], v1 = V[base + 3600], v2 = V[base + 7200],
            v3 = V[base + 10800];
      float sq  = ssq[b * 3600 + l];
      float sum = v0;                               // ones-column dot = sum(y)
      float var = (sq - sum * sum * (1.0f / 75.0f)) * (1.0f / 74.0f);
      float quad = xi[0]  * v0 * v0 + xi[5]  * v1 * v1 +
                   xi[10] * v2 * v2 + xi[15] * v3 * v3 +
                   2.f * (xi[1] * v0 * v1 + xi[2] * v0 * v2 + xi[3] * v0 * v3 +
                          xi[6] * v1 * v2 + xi[7] * v1 * v3 + xi[11] * v2 * v3);
      float err = (sq - quad) / (75.0f * var + 1e-20f);
      accp += (expf(-err) - c1) / (1.0f - c1) - 0.5f;
    }
  h1p[idx] = accp * 0.25f;
}

// ---------------------------------------------------------------------------
// Training-mode BN statistics: one block per channel. stats[c]=mean,
// stats[C+c]=rsqrt(var_biased+eps). Element at base + b*bS + c*cS + i.
// ---------------------------------------------------------------------------
__global__ void k_bn_stats(const float* __restrict__ in, float* __restrict__ stats,
                           int C, int nb, long long bStride, long long cStride,
                           int inner) {
  __shared__ float s1[256], s2[256];
  int c = blockIdx.x, t = threadIdx.x;
  float a1 = 0.f, a2 = 0.f;
  for (int b = 0; b < nb; ++b) {
    const float* p = in + (long long)b * bStride + (long long)c * cStride;
    for (int i = t; i < inner; i += 256) {
      float v = p[i]; a1 += v; a2 += v * v;
    }
  }
  s1[t] = a1; s2[t] = a2; __syncthreads();
  for (int ofs = 128; ofs > 0; ofs >>= 1) {
    if (t < ofs) { s1[t] += s1[t + ofs]; s2[t] += s2[t + ofs]; }
    __syncthreads();
  }
  if (t == 0) {
    float cnt  = (float)(nb * inner);
    float mean = s1[0] / cnt;
    float var  = s2[0] / cnt - mean * mean;
    stats[c] = mean; stats[C + c] = rsqrtf(var + BN_EPS);
  }
}

__global__ void k_bn_apply(float* __restrict__ data, const float* __restrict__ stats,
                           const float* __restrict__ g, const float* __restrict__ bb,
                           int C, int hw, int total) {
  int idx = blockIdx.x * blockDim.x + threadIdx.x;
  if (idx >= total) return;
  int c = (idx / hw) % C;
  data[idx] = (data[idx] - stats[c]) * stats[C + c] * g[c] + bb[c];
}

// ---------------------------------------------------------------------------
// im2col 3x3 VALID: f32 (B,Cin,Hin,Hin) -> packed bf16 B-operand
// [Cin*9][B*Hout*Hout]. Row ordering (c,kh,kw) matches OIHW weight reshape.
// ---------------------------------------------------------------------------
__global__ void k_im2col3x3(const float* __restrict__ in, bf16_t* __restrict__ out,
                            int Cin, int Hin, int Hout, int Bn) {
  long long idx = (long long)blockIdx.x * blockDim.x + threadIdx.x;
  int cols = Bn * Hout * Hout;
  long long total = (long long)Cin * 9 * cols;
  if (idx >= total) return;
  int col = (int)(idx % cols);
  int r   = (int)(idx / cols);
  int c = r / 9, kk = r % 9, kh = kk / 3, kw = kk % 3;
  int b = col / (Hout * Hout), p = col % (Hout * Hout);
  int oh = p / Hout, ow = p % Hout;
  float v = in[(((size_t)b * Cin + c) * Hin + oh + kh) * Hin + ow + kw];
  out[boff(r, col, cols)] = f2bf(v);
}

// Pack f32 weights (M x K row-major, OIHW-contiguous) into A-fragment layout.
__global__ void k_packA(const float* __restrict__ in, bf16_t* __restrict__ out,
                        int M, int K) {
  long long idx = (long long)blockIdx.x * blockDim.x + threadIdx.x;
  if (idx >= (long long)M * K) return;
  int m = (int)(idx / K), k = (int)(idx % K);
  out[aoff(m, k, K)] = f2bf(in[idx]);
}

// ---------------------------------------------------------------------------
// Fused BN -> ReLU -> s x s pool (avg or max). conv layout [C][Bn*Hin*Hin],
// column = b*HW + h*Hin + w. Output (Bn, C, Hin/s, Hin/s) row-major.
// ---------------------------------------------------------------------------
__global__ void k_bnrelu_pool(const float* __restrict__ conv,
                              const float* __restrict__ stats,
                              const float* __restrict__ g,
                              const float* __restrict__ bb,
                              float* __restrict__ out,
                              int C, int Hin, int Bn, int s, int isMax) {
  int Ho = Hin / s;
  int total = Bn * C * Ho * Ho;
  int idx = blockIdx.x * blockDim.x + threadIdx.x;
  if (idx >= total) return;
  int ow = idx % Ho; int t2 = idx / Ho;
  int oh = t2 % Ho;  t2 /= Ho;
  int c  = t2 % C;   int b = t2 / C;
  int HW = Hin * Hin;
  float mean = stats[c], is_ = stats[C + c], gg = g[c], bv = bb[c];
  float acc = isMax ? -1e30f : 0.f;
  for (int dy = 0; dy < s; ++dy)
    for (int dx = 0; dx < s; ++dx) {
      int col = b * HW + (oh * s + dy) * Hin + (ow * s + dx);
      float v = conv[(size_t)c * ((size_t)Bn * HW) + col];
      v = fmaxf((v - mean) * is_ * gg + bv, 0.f);
      acc = isMax ? fmaxf(acc, v) : acc + v;
    }
  out[idx] = isMax ? acc : acc / (float)(s * s);
}

__global__ void k_fc(const float* __restrict__ h, const float* __restrict__ w,
                     const float* __restrict__ bias, float* __restrict__ out) {
  int t = threadIdx.x;
  if (t >= BATCH * 5) return;
  int b = t / 5, cls = t % 5;
  float acc = bias[cls];
  for (int c = 0; c < 128; ++c) acc += h[b * 128 + c] * w[cls * 128 + c];
  out[t] = acc;
}

// ---------------------------------------------------------------------------
extern "C" void kernel_launch(void* const* d_in, const int* in_sizes, int n_in,
                              void* d_out, int out_size, void* d_ws, size_t ws_size,
                              hipStream_t stream) {
  (void)in_sizes; (void)n_in; (void)out_size;
  const float* x       = (const float*)d_in[0];
  const float* srn_w   = (const float*)d_in[1];
  const float* conv2_w = (const float*)d_in[2];   // conv*_b cancel under BN
  const float* conv3_w = (const float*)d_in[4];
  const float* conv4_w = (const float*)d_in[6];
  const float* bn1_g = (const float*)d_in[8],  * bn1_b = (const float*)d_in[9];
  const float* bn2_g = (const float*)d_in[10], * bn2_b = (const float*)d_in[11];
  const float* bn3_g = (const float*)d_in[12], * bn3_b = (const float*)d_in[13];
  const float* bn4_g = (const float*)d_in[14], * bn4_b = (const float*)d_in[15];
  const float* fc2_w = (const float*)d_in[16], * fc2_b = (const float*)d_in[17];
  float* out = (float*)d_out;

  char* ws = (char*)d_ws;
  size_t off = 0;
  auto alloc = [&](size_t bytes) -> char* {
    char* p = ws + off;
    off = (off + bytes + 255) & ~(size_t)255;
    return p;
  };
  bf16_t* ybf  = (bf16_t*)alloc((size_t)BATCH * 96 * 3600 * 2);
  float*  ssq  = (float*) alloc((size_t)BATCH * 3600 * 4);
  bf16_t* Xe   = (bf16_t*)alloc((size_t)256 * 96 * 2);
  float*  Xinv = (float*) alloc((size_t)64 * 16 * 4);
  float*  V    = (float*) alloc((size_t)BATCH * 256 * 3600 * 4);
  float*  h1p  = (float*) alloc((size_t)BATCH * 64 * 900 * 4);
  float*  st1  = (float*) alloc(2 * 128 * 4);
  float*  st2  = (float*) alloc(2 * 128 * 4);
  float*  st3  = (float*) alloc(2 * 128 * 4);
  float*  st4  = (float*) alloc(2 * 128 * 4);
  bf16_t* xin2 = (bf16_t*)alloc((size_t)576 * 3136 * 2);
  bf16_t* Aw2  = (bf16_t*)alloc((size_t)128 * 576 * 2);
  float*  c2   = (float*) alloc((size_t)128 * 3136 * 4);
  float*  h2p  = (float*) alloc((size_t)BATCH * 128 * 196 * 4);
  bf16_t* xin3 = (bf16_t*)alloc((size_t)1152 * 576 * 2);
  bf16_t* Aw3  = (bf16_t*)alloc((size_t)128 * 1152 * 2);
  float*  c3   = (float*) alloc((size_t)128 * 576 * 4);
  float*  h3p  = (float*) alloc((size_t)BATCH * 128 * 36 * 4);
  bf16_t* xin4 = (bf16_t*)alloc((size_t)1152 * 64 * 2);
  bf16_t* Aw4  = (bf16_t*)alloc((size_t)128 * 1152 * 2);
  float*  c4   = (float*) alloc((size_t)128 * 64 * 4);
  float*  h4   = (float*) alloc((size_t)BATCH * 128 * 4);
  if (off > ws_size) return;

  // --- SRN: prep, unfold, v = X^T y as a 256x96x3600 bf16 WMMA GEMM ---
  k_prep_x<<<1, 256, 0, stream>>>(srn_w, Xe, Xinv);
  k_unfold<<<(BATCH * 3600 + 255) / 256, 256, 0, stream>>>(x, ybf, ssq);
  k_gemm_bf16<<<dim3(225, 4, BATCH), 32, 0, stream>>>(
      (const v16bf*)Xe, (const v16bf*)ybf, V, 256, 3600, 96,
      (96LL * 3600) / 16, 256LL * 3600);
  k_srn_post<<<(BATCH * 64 * 900 + 255) / 256, 256, 0, stream>>>(V, ssq, Xinv, h1p);

  // --- BN1 (post-pool) then conv2 as WMMA im2col GEMM ---
  k_bn_stats<<<64, 256, 0, stream>>>(h1p, st1, 64, BATCH, 64LL * 900, 900LL, 900);
  k_bn_apply<<<(BATCH * 64 * 900 + 255) / 256, 256, 0, stream>>>(
      h1p, st1, bn1_g, bn1_b, 64, 900, BATCH * 64 * 900);
  k_packA<<<(73728 + 255) / 256, 256, 0, stream>>>(conv2_w, Aw2, 128, 576);
  k_im2col3x3<<<(int)(((long long)576 * 3136 + 255) / 256), 256, 0, stream>>>(
      h1p, xin2, 64, 30, 28, BATCH);
  k_gemm_bf16<<<dim3(196, 2, 1), 32, 0, stream>>>(
      (const v16bf*)Aw2, (const v16bf*)xin2, c2, 128, 3136, 576, 0, 0);
  k_bn_stats<<<128, 256, 0, stream>>>(c2, st2, 128, 1, 0, 3136LL, 3136);
  k_bnrelu_pool<<<(BATCH * 128 * 196 + 255) / 256, 256, 0, stream>>>(
      c2, st2, bn2_g, bn2_b, h2p, 128, 28, BATCH, 2, 0);

  // --- conv3 ---
  k_packA<<<(147456 + 255) / 256, 256, 0, stream>>>(conv3_w, Aw3, 128, 1152);
  k_im2col3x3<<<(int)(((long long)1152 * 576 + 255) / 256), 256, 0, stream>>>(
      h2p, xin3, 128, 14, 12, BATCH);
  k_gemm_bf16<<<dim3(36, 2, 1), 32, 0, stream>>>(
      (const v16bf*)Aw3, (const v16bf*)xin3, c3, 128, 576, 1152, 0, 0);
  k_bn_stats<<<128, 256, 0, stream>>>(c3, st3, 128, 1, 0, 576LL, 576);
  k_bnrelu_pool<<<(BATCH * 128 * 36 + 255) / 256, 256, 0, stream>>>(
      c3, st3, bn3_g, bn3_b, h3p, 128, 12, BATCH, 2, 1);

  // --- conv4 ---
  k_packA<<<(147456 + 255) / 256, 256, 0, stream>>>(conv4_w, Aw4, 128, 1152);
  k_im2col3x3<<<(int)(((long long)1152 * 64 + 255) / 256), 256, 0, stream>>>(
      h3p, xin4, 128, 6, 4, BATCH);
  k_gemm_bf16<<<dim3(4, 2, 1), 32, 0, stream>>>(
      (const v16bf*)Aw4, (const v16bf*)xin4, c4, 128, 64, 1152, 0, 0);
  k_bn_stats<<<128, 256, 0, stream>>>(c4, st4, 128, 1, 0, 64LL, 64);
  k_bnrelu_pool<<<(BATCH * 128 + 255) / 256, 256, 0, stream>>>(
      c4, st4, bn4_g, bn4_b, h4, 128, 4, BATCH, 4, 1);

  // --- FC head ---
  k_fc<<<1, 32, 0, stream>>>(h4, fc2_w, fc2_b, out);
}
